// GLABlock_2113123909866
// MI455X (gfx1250) — compile-verified, matrix operand order checked by reading, not run on hardware
//
#include <hip/hip_runtime.h>
#include <hip/hip_bf16.h>
#include <math.h>

// ---------------------------------------------------------------------------
// GLA block for MI455X (gfx1250, wave32, WMMA).
//   1) proj_kernel      : fused GEMM  x@[Wq|Wk|Wv|Wg]  (f16 WMMA, f32 accum)
//   2) lr1/lr2 kernels  : low-rank gate  gk = logsigmoid((x@Wgk1)@Wgk2+b)/16
//   3) gla_scan_kernel  : chunked gated scan (chunk=64), q-hat/k-hat form,
//                         all tiles laid out so WMMA fragments are 2x b128
//   4) epilogue_kernel  : per-head RMSNorm + swish output gate
//   5) out_gemm_kernel  : o @ Wo (f16 WMMA)
// ---------------------------------------------------------------------------

typedef __attribute__((ext_vector_type(16))) _Float16 v16h;
typedef __attribute__((ext_vector_type(8)))  _Float16 v8h;
typedef __attribute__((ext_vector_type(4)))  _Float16 v4h;
typedef __attribute__((ext_vector_type(8)))  float    v8f;
typedef __attribute__((ext_vector_type(4)))  float    v4f;

constexpr int Bc  = 4;
constexpr int Tc  = 2048;
constexpr int HIDc = 512;
constexpr int Hc  = 4;
constexpr int DKc = 64;
constexpr int DVc = 128;
constexpr int CH  = 64;              // scan chunk length
constexpr int TOK = Bc * Tc;         // 8192 tokens
constexpr float RMS_EPS = 1e-5f;

__device__ __forceinline__ v8f wmma16(v16h a, v16h b, v8f c) {
  return __builtin_amdgcn_wmma_f32_16x16x32_f16(false, a, false, b, (short)0, c,
                                                false, false);
}

// 16-bit fragment (A role: row=M; B role with transposed storage: row=N).
// Lane layout (ISA 7.12.2): lanes 0-15 hold K={0..7,16..23}, lanes 16-31 hold
// K={8..15,24..31}  ->  two contiguous 16B chunks per lane => 2x ds_load_b128.
// Requires: base 16B-aligned, ld (in halves) a multiple of 8.
__device__ __forceinline__ v16h frag16(const _Float16* p, int ld, int rowBase,
                                       int kBase) {
  const int lane = threadIdx.x & 31, half = lane >> 4;
  const _Float16* row = p + (size_t)(rowBase + (lane & 15)) * ld + kBase + half * 8;
  v8h lo = *(const v8h*)(row);
  v8h hi = *(const v8h*)(row + 16);
  return __builtin_shufflevector(lo, hi, 0, 1, 2, 3, 4, 5, 6, 7,
                                 8, 9, 10, 11, 12, 13, 14, 15);
}

// Same fragment gathered from fp32 LDS (state), cvt -> f16. ld multiple of 4.
__device__ __forceinline__ v16h frag16_f32(const float* p, int ld, int rowBase,
                                           int kBase) {
  const int lane = threadIdx.x & 31, half = lane >> 4;
  const float* row = p + (size_t)(rowBase + (lane & 15)) * ld + kBase + half * 8;
  v4f a0 = *(const v4f*)(row);
  v4f a1 = *(const v4f*)(row + 4);
  v4f b0 = *(const v4f*)(row + 16);
  v4f b1 = *(const v4f*)(row + 20);
  v16h f;
#pragma unroll
  for (int e = 0; e < 4; ++e) {
    f[e]      = (_Float16)a0[e];
    f[4 + e]  = (_Float16)a1[e];
    f[8 + e]  = (_Float16)b0[e];
    f[12 + e] = (_Float16)b1[e];
  }
  return f;
}

// ---------------------------------------------------------------------------
// 1) Fused projection GEMM: [8192,512] @ [512,1536] -> q(256)|k(256)|v(512)|g(512)
//    128x128 block tile, 8 waves, 8 WMMA per K-step per wave.
// ---------------------------------------------------------------------------
__global__ __launch_bounds__(256) void proj_kernel(
    const float* __restrict__ x,
    const float* __restrict__ Wq, const float* __restrict__ Wk,
    const float* __restrict__ Wv, const float* __restrict__ Wg,
    float* __restrict__ qws, float* __restrict__ kws,
    float* __restrict__ vws, float* __restrict__ gws) {
  __shared__ __align__(16) _Float16 As[128][40];    // A[m][k], k contiguous
  __shared__ __align__(16) _Float16 BsT[128][40];   // B^T[n][k], k contiguous
  const int m0 = blockIdx.x * 128, n0 = blockIdx.y * 128;
  const int tid = threadIdx.x, lane = tid & 31, wave = tid >> 5;
  const int mr0 = (wave & 3) * 32;        // two 16-row tiles
  const int nc0 = (wave >> 2) * 64;       // four 16-col tiles
  v8f acc[2][4] = {};
  for (int kk = 0; kk < HIDc; kk += 32) {
    for (int i = tid; i < 1024; i += 256) {          // A: 128x32 via float4
      const int row = i >> 3, cg = i & 7;
      v4f xv = *(const v4f*)(x + (size_t)(m0 + row) * HIDc + kk + cg * 4);
      v4h hv;
#pragma unroll
      for (int e = 0; e < 4; ++e) hv[e] = (_Float16)xv[e];
      *(v4h*)&As[row][cg * 4] = hv;
    }
    for (int i = tid; i < 1024; i += 256) {          // B: 32x128, store transposed
      const int k = i >> 5, ng = i & 31, n = n0 + ng * 4;
      v4f wv;
      if (n < 256)       wv = *(const v4f*)(Wq + (size_t)(kk + k) * 256 + n);
      else if (n < 512)  wv = *(const v4f*)(Wk + (size_t)(kk + k) * 256 + (n - 256));
      else if (n < 1024) wv = *(const v4f*)(Wv + (size_t)(kk + k) * 512 + (n - 512));
      else               wv = *(const v4f*)(Wg + (size_t)(kk + k) * 512 + (n - 1024));
#pragma unroll
      for (int j2 = 0; j2 < 4; ++j2) BsT[ng * 4 + j2][k] = (_Float16)wv[j2];
    }
    if (kk + 32 < HIDc)
      __builtin_prefetch(&x[(size_t)(m0 + (tid >> 1)) * HIDc + kk + 32], 0, 1);
    __syncthreads();
    v16h a0 = frag16(&As[0][0], 40, mr0, 0);
    v16h a1 = frag16(&As[0][0], 40, mr0 + 16, 0);
#pragma unroll
    for (int j = 0; j < 4; ++j) {
      v16h bm = frag16(&BsT[0][0], 40, nc0 + j * 16, 0);
      acc[0][j] = wmma16(a0, bm, acc[0][j]);
      acc[1][j] = wmma16(a1, bm, acc[1][j]);
    }
    __syncthreads();
  }
  const int nl = lane & 15, mOff = (lane >> 4) * 8;
#pragma unroll
  for (int a2 = 0; a2 < 2; ++a2) {
#pragma unroll
    for (int j = 0; j < 4; ++j) {
      const int n = n0 + nc0 + j * 16 + nl;
#pragma unroll
      for (int r = 0; r < 8; ++r) {
        const size_t m = (size_t)(m0 + mr0 + a2 * 16 + mOff + r);
        const float val = acc[a2][j][r];
        if (n < 256)       qws[m * 256 + n]          = val * 0.125f;  // DK^-0.5
        else if (n < 512)  kws[m * 256 + (n - 256)]  = val;
        else if (n < 1024) vws[m * 512 + (n - 512)]  = val;
        else               gws[m * 512 + (n - 1024)] = val;
      }
    }
  }
}

// ---------------------------------------------------------------------------
// 2) Low-rank gate path (tiny; VALU)
// ---------------------------------------------------------------------------
__global__ __launch_bounds__(256) void lr1_kernel(const float* __restrict__ x,
                                                  const float* __restrict__ Wgk1,
                                                  float* __restrict__ xg1) {
  __shared__ float xs[16][512];
  const int t0 = blockIdx.x * 16;
  for (int i = threadIdx.x; i < 16 * 512; i += 256)
    xs[i >> 9][i & 511] = x[(size_t)(t0 + (i >> 9)) * 512 + (i & 511)];
  __syncthreads();
  const int tok = threadIdx.x >> 4, j = threadIdx.x & 15;
  float s = 0.f;
  for (int k2 = 0; k2 < 512; ++k2) s += xs[tok][k2] * Wgk1[k2 * 16 + j];
  xg1[(size_t)(t0 + tok) * 16 + j] = s;
}

__global__ __launch_bounds__(256) void lr2_kernel(const float* __restrict__ xg1,
                                                  const float* __restrict__ Wgk2,
                                                  const float* __restrict__ bgk2,
                                                  float* __restrict__ gk) {
  const int idx = blockIdx.x * 256 + threadIdx.x;  // TOK*256
  const int tok = idx >> 8, n = idx & 255;
  float s = bgk2[n];
#pragma unroll
  for (int j = 0; j < 16; ++j) s += xg1[(size_t)tok * 16 + j] * Wgk2[j * 256 + n];
  const float ls = fminf(s, 0.f) - log1pf(__expf(-fabsf(s)));  // stable logsigmoid
  gk[idx] = ls * (1.f / 16.f);
}

__global__ void zero_kernel(float* __restrict__ p, int n) {
  const int i = blockIdx.x * 256 + threadIdx.x;
  if (i < n) p[i] = 0.f;
}

// ---------------------------------------------------------------------------
// 3) Chunked GLA scan. grid=(B*H, 2): one (b,h) per block, DV split in two
//    64-wide halves. q-hat/k-hat form:
//      qh = q*exp(L_t - L_C)  (clamped),  kh = k*exp(L_C - L_t)  (bounded)
//      ST <- exp(L_C) (.) ST           (state pre-decay, in place)
//      A  = causal(qh kh^T)            (WMMA)
//      o  = A @ v + qh @ ST            (WMMA)
//      ST += kh^T @ v                  (WMMA)
//    Every f16 tile keeps the contraction dim contiguous -> 2x b128 frags.
// ---------------------------------------------------------------------------
__global__ __launch_bounds__(256) void gla_scan_kernel(
    const float* __restrict__ qg, const float* __restrict__ kg,
    const float* __restrict__ vg, const float* __restrict__ gkg,
    float* __restrict__ on, float* __restrict__ ssq_g) {
  constexpr int LDA = CH + 8;   // 72 halves = 144 B pitch (16B aligned)
  constexpr int LDS_ = 68;      // 68 floats = 272 B pitch (16B aligned)
  __shared__ __align__(16) float    ST[64][LDS_];   // state^T [n][d] fp32
  __shared__ __align__(16) _Float16 qt[CH][LDA];    // qh [t][d]
  __shared__ __align__(16) _Float16 kt[CH][LDA];    // kh [t][d]   (B side)
  __shared__ __align__(16) _Float16 ktT[DKc][LDA];  // kh^T [d][t] (A side)
  __shared__ __align__(16) _Float16 Am[CH][LDA];    // A  [t][s]
  __shared__ __align__(16) _Float16 vT[64][LDA];    // v^T [n][s]
  __shared__ float expLC[DKc];
  __shared__ float ssql[CH];

  const int bh = blockIdx.x, b = bh >> 2, h = bh & 3;
  const int nHalf = blockIdx.y;
  const int tid = threadIdx.x, lane = tid & 31, wave = tid >> 5;

  for (int i = tid; i < 64 * LDS_; i += 256) (&ST[0][0])[i] = 0.f;
  __syncthreads();

  for (int c = 0; c < Tc / CH; ++c) {
    const int t0 = c * CH;
    const float* qb  = qg  + ((size_t)(b * Tc + t0) * Hc + h) * DKc;
    const float* kb  = kg  + ((size_t)(b * Tc + t0) * Hc + h) * DKc;
    const float* gkb = gkg + ((size_t)(b * Tc + t0) * Hc + h) * DKc;
    const float* vb  = vg  + (size_t)(b * Tc + t0) * (Hc * DVc) + h * DVc + nHalf * 64;
    float*       ob  = on  + (size_t)(b * Tc + t0) * (Hc * DVc) + h * DVc + nHalf * 64;

    // ---- step 1: gate cumsum; build qh, kh (+ transposed copy) ----
    if (tid < DKc) {
      const int d = tid;
      float lamC = 0.f;
      for (int t = 0; t < CH; ++t) lamC += gkb[(size_t)t * (Hc * DKc) + d];
      expLC[d] = __expf(lamC);
      float lam = 0.f;
      for (int t = 0; t < CH; ++t) {
        const size_t off = (size_t)t * (Hc * DKc) + d;
        lam += gkb[off];
        const float eq = __expf(lam - lamC);               // >= 1
        const float qh = fmaxf(fminf(qb[off] * eq, 60000.f), -60000.f);
        const float kh = kb[off] / eq;                     // bounded by |k|
        qt[t][d]  = (_Float16)qh;
        kt[t][d]  = (_Float16)kh;
        ktT[d][t] = (_Float16)kh;
      }
    }
    if (tid < CH) ssql[tid] = 0.f;
    __syncthreads();

    // ---- step 1.5: pre-decay state in place; stage v^T ----
    for (int i = tid; i < 64 * 64; i += 256) {
      const int n = i >> 6, d = i & 63;
      ST[n][d] *= expLC[d];
    }
    for (int i = tid; i < CH * 64; i += 256) {
      const int s = i >> 6, n = i & 63;
      vT[n][s] = (_Float16)vb[(size_t)s * (Hc * DVc) + n];
    }

    // ---- step 2: A = causal(qh kh^T), 4x4 tiles, 2 per wave ----
    {
      const int tRow = wave & 3, sp0 = (wave >> 2) * 2;
#pragma unroll
      for (int spi = 0; spi < 2; ++spi) {
        const int sTile = sp0 + spi;
        v8f acc = {};
        if (sTile <= tRow) {                 // wave-uniform; EXEC stays all-1s
#pragma unroll
          for (int k2 = 0; k2 < 2; ++k2) {
            v16h a  = frag16(&qt[0][0], LDA, tRow * 16, k2 * 32);
            v16h bm = frag16(&kt[0][0], LDA, sTile * 16, k2 * 32);  // B[d][s]
            acc = wmma16(a, bm, acc);
          }
        }
        const int sIdx = sTile * 16 + (lane & 15), mOff = (lane >> 4) * 8;
#pragma unroll
        for (int r = 0; r < 8; ++r) {
          const int tIdx = tRow * 16 + mOff + r;
          Am[tIdx][sIdx] = (_Float16)((sIdx <= tIdx) ? acc[r] : 0.f);
        }
      }
    }
    __syncthreads();

    // ---- step 3: o = A@v + qh@ST ; raw o out, sum-of-squares via LDS atomics ----
    {
      const int tRow = wave & 3, nt0 = (wave >> 2) * 2;
      v8f acc[2] = {};
#pragma unroll
      for (int k2 = 0; k2 < 2; ++k2) {
        v16h aA = frag16(&Am[0][0], LDA, tRow * 16, k2 * 32);
        v16h aQ = frag16(&qt[0][0], LDA, tRow * 16, k2 * 32);
#pragma unroll
        for (int j = 0; j < 2; ++j) {
          v16h bv = frag16(&vT[0][0], LDA, (nt0 + j) * 16, k2 * 32);
          acc[j] = wmma16(aA, bv, acc[j]);
          v16h bs = frag16_f32(&ST[0][0], LDS_, (nt0 + j) * 16, k2 * 32);
          acc[j] = wmma16(aQ, bs, acc[j]);
        }
      }
      const int nl = lane & 15, mOff = (lane >> 4) * 8;
#pragma unroll
      for (int j = 0; j < 2; ++j) {
        const int n = (nt0 + j) * 16 + nl;
#pragma unroll
        for (int r = 0; r < 8; ++r) {
          const int t = tRow * 16 + mOff + r;
          ob[(size_t)t * (Hc * DVc) + n] = acc[j][r];
          atomicAdd(&ssql[t], acc[j][r] * acc[j][r]);
        }
      }
    }
    __syncthreads();  // all ST reads done; ssql complete
    if (tid < CH)
      atomicAdd(&ssq_g[(size_t)(b * Tc + t0 + tid) * Hc + h], ssql[tid]);

    // ---- step 4: ST += kh^T @ v ----
    {
      const int dRow = wave & 3, nt0 = (wave >> 2) * 2;
      v8f acc[2] = {};
#pragma unroll
      for (int k2 = 0; k2 < 2; ++k2) {
        v16h aK = frag16(&ktT[0][0], LDA, dRow * 16, k2 * 32);
#pragma unroll
        for (int j = 0; j < 2; ++j) {
          v16h bv = frag16(&vT[0][0], LDA, (nt0 + j) * 16, k2 * 32);
          acc[j] = wmma16(aK, bv, acc[j]);
        }
      }
      const int nl = lane & 15, mOff = (lane >> 4) * 8;
#pragma unroll
      for (int j = 0; j < 2; ++j) {
        const int n = (nt0 + j) * 16 + nl;
#pragma unroll
        for (int r = 0; r < 8; ++r) {
          const int d = dRow * 16 + mOff + r;
          ST[n][d] += acc[j][r];
        }
      }
    }
    __syncthreads();  // protect qt/kt/ktT/Am/vT/ST before next chunk
  }
}

// ---------------------------------------------------------------------------
// 4) RMSNorm + swish output gate, in place on raw o
// ---------------------------------------------------------------------------
__global__ __launch_bounds__(256) void epilogue_kernel(
    float* __restrict__ on, const float* __restrict__ g,
    const float* __restrict__ rms_w, const float* __restrict__ ssq) {
  const size_t idx = (size_t)blockIdx.x * 256 + threadIdx.x;  // TOK*512
  const size_t tok = idx >> 9;
  const int col = (int)(idx & 511), h = col >> 7, n = col & 127;
  const float ms = ssq[tok * Hc + h] * (1.f / DVc);
  const float rstd = rsqrtf(ms + RMS_EPS);
  const float gg = g[idx];
  const float val = on[idx] * rstd * rms_w[n];
  on[idx] = val * gg * (1.f / (1.f + __expf(-gg)));
}

// ---------------------------------------------------------------------------
// 5) Output projection: [8192,512] @ [512,512], 128x128 block tile
// ---------------------------------------------------------------------------
__global__ __launch_bounds__(256) void out_gemm_kernel(
    const float* __restrict__ A, const float* __restrict__ Bw,
    float* __restrict__ Cm) {
  __shared__ __align__(16) _Float16 As[128][40];
  __shared__ __align__(16) _Float16 BsT[128][40];
  const int m0 = blockIdx.x * 128, n0 = blockIdx.y * 128;
  const int tid = threadIdx.x, lane = tid & 31, wave = tid >> 5;
  const int mr0 = (wave & 3) * 32, nc0 = (wave >> 2) * 64;
  v8f acc[2][4] = {};
  for (int kk = 0; kk < 512; kk += 32) {
    for (int i = tid; i < 1024; i += 256) {
      const int row = i >> 3, cg = i & 7;
      v4f xv = *(const v4f*)(A + (size_t)(m0 + row) * 512 + kk + cg * 4);
      v4h hv;
#pragma unroll
      for (int e = 0; e < 4; ++e) hv[e] = (_Float16)xv[e];
      *(v4h*)&As[row][cg * 4] = hv;
    }
    for (int i = tid; i < 1024; i += 256) {
      const int k = i >> 5, ng = i & 31;
      v4f wv = *(const v4f*)(Bw + (size_t)(kk + k) * 512 + n0 + ng * 4);
#pragma unroll
      for (int j2 = 0; j2 < 4; ++j2) BsT[ng * 4 + j2][k] = (_Float16)wv[j2];
    }
    if (kk + 32 < 512)
      __builtin_prefetch(&A[(size_t)(m0 + (tid >> 1)) * 512 + kk + 32], 0, 1);
    __syncthreads();
    v16h a0 = frag16(&As[0][0], 40, mr0, 0);
    v16h a1 = frag16(&As[0][0], 40, mr0 + 16, 0);
#pragma unroll
    for (int j = 0; j < 4; ++j) {
      v16h bm = frag16(&BsT[0][0], 40, nc0 + j * 16, 0);
      acc[0][j] = wmma16(a0, bm, acc[0][j]);
      acc[1][j] = wmma16(a1, bm, acc[1][j]);
    }
    __syncthreads();
  }
  const int nl = lane & 15, mOff = (lane >> 4) * 8;
#pragma unroll
  for (int a2 = 0; a2 < 2; ++a2) {
#pragma unroll
    for (int j = 0; j < 4; ++j) {
      const int n = n0 + nc0 + j * 16 + nl;
#pragma unroll
      for (int r = 0; r < 8; ++r)
        Cm[(size_t)(m0 + mr0 + a2 * 16 + mOff + r) * 512 + n] = acc[a2][j][r];
    }
  }
}

// ---------------------------------------------------------------------------
extern "C" void kernel_launch(void* const* d_in, const int* in_sizes, int n_in,
                              void* d_out, int out_size, void* d_ws, size_t ws_size,
                              hipStream_t stream) {
  (void)in_sizes; (void)n_in; (void)out_size; (void)ws_size;
  const float* x    = (const float*)d_in[0];
  const float* Wq   = (const float*)d_in[1];
  const float* Wk   = (const float*)d_in[2];
  const float* Wv   = (const float*)d_in[3];
  const float* Wg   = (const float*)d_in[4];
  const float* Wo   = (const float*)d_in[5];
  const float* Wgk1 = (const float*)d_in[6];
  const float* Wgk2 = (const float*)d_in[7];
  const float* bgk2 = (const float*)d_in[8];
  const float* rmsw = (const float*)d_in[9];
  float* out = (float*)d_out;

  float* ws   = (float*)d_ws;
  float* qws  = ws;                              // [TOK,256]
  float* kws  = qws  + (size_t)TOK * 256;        // [TOK,256]
  float* vws  = kws  + (size_t)TOK * 256;        // [TOK,512]
  float* gws  = vws  + (size_t)TOK * 512;        // [TOK,512]
  float* gkws = gws  + (size_t)TOK * 512;        // [TOK,256]
  float* xg1  = gkws + (size_t)TOK * 256;        // [TOK,16]
  float* on   = xg1  + (size_t)TOK * 16;         // [TOK,512] raw o -> gated o
  float* ssq  = on   + (size_t)TOK * 512;        // [TOK,H]

  proj_kernel<<<dim3(TOK / 128, 1536 / 128), 256, 0, stream>>>(
      x, Wq, Wk, Wv, Wg, qws, kws, vws, gws);
  lr1_kernel<<<TOK / 16, 256, 0, stream>>>(x, Wgk1, xg1);
  lr2_kernel<<<(TOK * 256) / 256, 256, 0, stream>>>(xg1, Wgk2, bgk2, gkws);
  zero_kernel<<<(TOK * Hc + 255) / 256, 256, 0, stream>>>(ssq, TOK * Hc);
  gla_scan_kernel<<<dim3(Bc * Hc, 2), 256, 0, stream>>>(
      qws, kws, vws, gkws, on, ssq);
  epilogue_kernel<<<(TOK * 512) / 256, 256, 0, stream>>>(on, gws, rmsw, ssq);
  out_gemm_kernel<<<dim3(TOK / 128, 512 / 128), 256, 0, stream>>>(on, Wo, out);
}
